// SAGELayer_84868553769261
// MI455X (gfx1250) — compile-verified, matrix operand order checked by reading, not run on hardware
//
#include <hip/hip_runtime.h>
#include <math.h>

// ---------------------------------------------------------------------------
// SAGE layer for MI455X (gfx1250, wave32, WMMA, async-LDS, double-buffered).
//   h      = relu(X @ W_w + b_w)        -> bf16, K-pair-interleaved (hP)
//   pooled = (A @ h) / (rowsum(A)+eps)  -> A streamed ONCE, rowsum fused
//   cat    = [X | pooled] (bf16, stride 512)
//   out    = relu(cat @ W_fc + b_fc) + eps ; out /= (||out||_F + eps)
// All GEMMs: v_wmma_f32_16x16x32_bf16.  h tiles staged to LDS with
// global_load_async_to_lds_b128 (ASYNCcnt); A tiles converted f32->bf16 via
// v_cvt_pk_bf16_f32; LDS double-buffered so copy/convert of tile i+1 overlaps
// the 16-WMMA compute of tile i (one barrier per K-step).
// ---------------------------------------------------------------------------

#define NN    8192
#define DIN   256
#define DOUT  256
#define DCAT  512
#define EPSF  1e-8f

typedef __attribute__((ext_vector_type(16))) __bf16 v16bf;
typedef __attribute__((ext_vector_type(8)))  float  v8f;

union FragBF {
    v16bf          v;
    unsigned short us[16];
    unsigned       u32[8];
};

// float -> bf16 native convert (RNE); packer for two floats -> one dword
__device__ __forceinline__ unsigned short f2bf(float f) {
    __bf16 b = (__bf16)f;
    return __builtin_bit_cast(unsigned short, b);
}
__device__ __forceinline__ unsigned pack2bf(float lo, float hi) {
    return ((unsigned)f2bf(hi) << 16) | (unsigned)f2bf(lo);
}

// K index of bf16 element e (0..15) of an A/B fragment, lane-group g (0/1).
// ISA 7.12.2: VGPR0..3 hold K = {0..7}+8g, VGPR4..7 hold K = {16..23}+8g.
__device__ __forceinline__ int ke(int e, int g) { return (e < 8 ? e : e + 8) + 8 * g; }
// K base of the dword-pair v (elements 2v, 2v+1)
__device__ __forceinline__ int kb(int v, int g) { return (v < 4 ? 2 * v : 8 + 2 * v) + 8 * g; }
// K-pair index of dword v: kb(v,g)/2
__device__ __forceinline__ int kp(int v, int g) { return (v < 4 ? v : v + 4) + 4 * g; }

__device__ __forceinline__ v8f wmma_bf16(const FragBF& a, const FragBF& b, v8f c) {
    return __builtin_amdgcn_wmma_f32_16x16x32_bf16(
        /*neg_a=*/false, a.v, /*neg_b=*/false, b.v,
        /*c_mod=*/(short)0, c, /*reuse_a=*/false, /*reuse_b=*/false);
}

// ---------------------------------------------------------------------------
// Kernel 1: h = relu(X @ W_w + b_w) -> hP, K-pair interleaved bf16:
//   hP[k2*256 + n] = { h[2*k2+1][n] , h[2*k2][n] }   (hi , lo)
// One 16x16 tile per wave; 1024 blocks x 8 waves.
// ---------------------------------------------------------------------------
__global__ void __launch_bounds__(256) k_h(const float* __restrict__ X,
                                           const float* __restrict__ Ww,
                                           const float* __restrict__ bw,
                                           unsigned* __restrict__ hP) {
    const int lane = threadIdx.x & 31, wid = threadIdx.x >> 5;
    const int tile = blockIdx.x * 8 + wid;
    const int m0 = (tile >> 4) * 16, n0 = (tile & 15) * 16;
    const int g = lane >> 4, ln = lane & 15;

    v8f acc = {0.f, 0.f, 0.f, 0.f, 0.f, 0.f, 0.f, 0.f};
    for (int k0 = 0; k0 < DIN; k0 += 32) {
        FragBF fa, fb;
        const int m = m0 + ln, n = n0 + ln;
#pragma unroll
        for (int v = 0; v < 8; ++v) {      // contiguous K pairs -> float2 loads
            const float2 p = *(const float2*)(X + (size_t)m * DIN + k0 + kb(v, g));
            fa.u32[v] = pack2bf(p.x, p.y);
        }
#pragma unroll
        for (int e = 0; e < 16; ++e)
            fb.us[e] = f2bf(Ww[(size_t)(k0 + ke(e, g)) * DOUT + n]);
        acc = wmma_bf16(fa, fb, acc);
    }
    const float bias = bw[n0 + ln];
#pragma unroll
    for (int i = 0; i < 4; ++i) {          // rows (m0+2i+8g, m0+2i+1+8g) pair up
        const float lo = fmaxf(acc[2 * i]     + bias, 0.f);
        const float hi = fmaxf(acc[2 * i + 1] + bias, 0.f);
        hP[(size_t)(m0 / 2 + i + 4 * g) * DOUT + (n0 + ln)] = pack2bf(lo, hi);
    }
}

// ---------------------------------------------------------------------------
// Kernel 2: cast X into cat[:, 0:256] as bf16 (cat row stride 512).
// ---------------------------------------------------------------------------
__global__ void k_castX(const float* __restrict__ X, unsigned short* __restrict__ catB) {
    size_t i = (size_t)blockIdx.x * blockDim.x + threadIdx.x;
    if (i < (size_t)NN * DIN) {
        size_t r = i >> 8;
        int    c = (int)(i & 255);
        catB[r * DCAT + c] = f2bf(X[i]);
    }
}

// ---------------------------------------------------------------------------
// Kernel 3 (dominant): pooled = (A @ h) / (rowsum(A)+eps) -> bf16 cat[:,256:512]
// Block = 128 rows x 256 cols, full K sweep => A streamed exactly once with
// fused row sums.  64 blocks x 256 thr (8 waves, 4(m) x 2(n) wave grid).
// Double-buffered LDS: async h-copy + A convert of tile i+1 overlap WMMA of
// tile i; single barrier per K-step.
// ---------------------------------------------------------------------------
__global__ void __launch_bounds__(256) k_pool(const float* __restrict__ A,
                                              const unsigned* __restrict__ hP,
                                              unsigned short* __restrict__ catB) {
    __shared__ unsigned short Asm[2][128][34]; // bf16 A tiles, padded rows (68 B)
    __shared__ unsigned       Bsm[2][16][256]; // h tiles: [k-pair][n], dword = 2 K
    __shared__ float          rs[128];         // row sums of A for this block

    const int t = threadIdx.x;
    const int lane = t & 31, wid = t >> 5;
    const int g = lane >> 4, ln = lane & 15;
    const int i0 = blockIdx.x * 128;
    const int mw = (wid >> 1) * 32;   // wave row offset in block tile
    const int nw = (wid & 1) * 128;   // wave col offset

    if (t < 128) rs[t] = 0.f;

    float rsum[4] = {0.f, 0.f, 0.f, 0.f};

    // ---- stage h tile (async to LDS, ASYNCcnt) ----
    auto stageB = [&](int k0, int buf) {
#pragma unroll
        for (int p = 0; p < 4; ++p) {
            const int q   = p * 256 + t;
            const int row = q >> 6;            // 0..15 (k-pair row)
            const int c4  = (q & 63) * 4;      // dword column, 16 B chunks
            const unsigned  ldsa = (unsigned)(uintptr_t)&Bsm[buf][row][c4];
            const unsigned* gp   = hP + (size_t)(k0 / 2 + row) * DOUT + c4;
            asm volatile("global_load_async_to_lds_b128 %0, %1, off"
                         :: "v"(ldsa), "v"(gp) : "memory");
        }
    };
    // ---- stage A tile (f32 b128 loads -> v_cvt_pk_bf16_f32 -> LDS), rowsum fused
    auto stageA = [&](int k0, int buf) {
#pragma unroll
        for (int p = 0; p < 4; ++p) {
            const int q = p * 256 + t;
            const int r = q >> 3;              // = 32*p + (t>>3)
            const int c = (q & 7) * 4;
            const float4 a = *(const float4*)(A + (size_t)(i0 + r) * NN + k0 + c);
            rsum[p] += a.x + a.y + a.z + a.w;
            *(unsigned*)&Asm[buf][r][c]     = pack2bf(a.x, a.y);
            *(unsigned*)&Asm[buf][r][c + 2] = pack2bf(a.z, a.w);
        }
    };

    v8f acc[2][8];
    const v8f vz = {0.f, 0.f, 0.f, 0.f, 0.f, 0.f, 0.f, 0.f};
#pragma unroll
    for (int mi = 0; mi < 2; ++mi)
#pragma unroll
        for (int ni = 0; ni < 8; ++ni) acc[mi][ni] = vz;

    // prologue: fill buffer 0
    stageB(0, 0);
    stageA(0, 0);
    asm volatile("s_wait_asynccnt 0x0" ::: "memory");
    __syncthreads();

    for (int k0 = 0; k0 < NN; k0 += 32) {
        const int pb = (k0 >> 5) & 1;

        // ---- kick off tile i+1 into the other buffer (overlaps WMMA below)
        if (k0 + 32 < NN) {
            stageB(k0 + 32, pb ^ 1);
            stageA(k0 + 32, pb ^ 1);
        }
        // ---- speculative prefetch of tile i+2 (global_prefetch_b8)
        if (k0 + 64 < NN) {
            const int r = t >> 3, c = (t & 7) * 4;
            __builtin_prefetch((const void*)(A + (size_t)(i0 + r) * NN + k0 + 64 + c), 0, 1);
        }

        // ---- WMMA on tile i: 2 A-frags x 8 B-frags -> 16 wmma per wave
        FragBF fa[2];
#pragma unroll
        for (int mi = 0; mi < 2; ++mi) {
            const int m = mw + mi * 16 + ln;
#pragma unroll
            for (int v = 0; v < 8; ++v)
                fa[mi].u32[v] = *(const unsigned*)&Asm[pb][m][kb(v, g)];
        }
#pragma unroll
        for (int ni = 0; ni < 8; ++ni) {
            FragBF fb;
            const int n = nw + ni * 16 + ln;
#pragma unroll
            for (int v = 0; v < 8; ++v)       // one contiguous b32 per dword
                fb.u32[v] = Bsm[pb][kp(v, g)][n];
#pragma unroll
            for (int mi = 0; mi < 2; ++mi)
                acc[mi][ni] = wmma_bf16(fa[mi], fb, acc[mi][ni]);
        }

        // ---- single barrier: my reads of buf[pb] done, everyone's writes to
        //      buf[pb^1] (async + ds) complete
        asm volatile("s_wait_asynccnt 0x0" ::: "memory");
        __syncthreads();
    }

    // ---- finish row sums (8 partial lanes per row)
#pragma unroll
    for (int p = 0; p < 4; ++p) atomicAdd(&rs[32 * p + (t >> 3)], rsum[p]);
    __syncthreads();

    // ---- epilogue: divide by (rowsum+eps), cast bf16, write cat[:,256:512]
#pragma unroll
    for (int mi = 0; mi < 2; ++mi)
#pragma unroll
        for (int ni = 0; ni < 8; ++ni)
#pragma unroll
            for (int v = 0; v < 8; ++v) {
                const int ml = mw + mi * 16 + v + 8 * g;   // C layout: M = vgpr + 8*grp
                const int n  = nw + ni * 16 + ln;          //           N = lane%16
                const float val = acc[mi][ni][v] / (rs[ml] + EPSF);
                catB[(size_t)(i0 + ml) * DCAT + DIN + n] = f2bf(val);
            }
}

// ---------------------------------------------------------------------------
// Kernel 4: zero the squared-norm accumulator.
// ---------------------------------------------------------------------------
__global__ void k_init(float* s) { if (threadIdx.x == 0 && blockIdx.x == 0) *s = 0.f; }

// ---------------------------------------------------------------------------
// Kernel 5: out = relu(cat @ W_fc + b_fc) + eps ; accumulate sum(out^2).
// One 16x16 tile per wave, K=512. Wave32 shuffle reduce + 1 atomic per wave.
// ---------------------------------------------------------------------------
__global__ void __launch_bounds__(256) k_out(const unsigned short* __restrict__ catB,
                                             const float* __restrict__ Wfc,
                                             const float* __restrict__ bfc,
                                             float* __restrict__ out,
                                             float* __restrict__ sumsq) {
    const int lane = threadIdx.x & 31, wid = threadIdx.x >> 5;
    const int tile = blockIdx.x * 8 + wid;
    const int m0 = (tile >> 4) * 16, n0 = (tile & 15) * 16;
    const int g = lane >> 4, ln = lane & 15;

    v8f acc = {0.f, 0.f, 0.f, 0.f, 0.f, 0.f, 0.f, 0.f};
    for (int k0 = 0; k0 < DCAT; k0 += 32) {
        FragBF fa, fb;
        const int m = m0 + ln, n = n0 + ln;
#pragma unroll
        for (int v = 0; v < 8; ++v)        // bf16 K-pairs are contiguous: b32
            fa.u32[v] = *(const unsigned*)&catB[(size_t)m * DCAT + k0 + kb(v, g)];
#pragma unroll
        for (int e = 0; e < 16; ++e)
            fb.us[e] = f2bf(Wfc[(size_t)(k0 + ke(e, g)) * DOUT + n]);
        acc = wmma_bf16(fa, fb, acc);
    }
    const float bias = bfc[n0 + ln];
    float sq = 0.f;
#pragma unroll
    for (int v = 0; v < 8; ++v) {
        float val = fmaxf(acc[v] + bias, 0.f) + EPSF;
        out[(size_t)(m0 + v + 8 * g) * DOUT + (n0 + ln)] = val;
        sq += val * val;
    }
#pragma unroll
    for (int o = 16; o > 0; o >>= 1) sq += __shfl_xor(sq, o, 32);
    if (lane == 0) atomicAdd(sumsq, sq);
}

// ---------------------------------------------------------------------------
// Kernel 6: out *= 1/(sqrt(sumsq)+eps).  524288 float4's.
// ---------------------------------------------------------------------------
__global__ void k_scale(float* __restrict__ out, const float* __restrict__ sumsq) {
    const float inv = 1.0f / (sqrtf(*sumsq) + EPSF);
    const size_t i = (size_t)blockIdx.x * blockDim.x + threadIdx.x;
    if (i < (size_t)NN * DOUT / 4) {
        float4* o4 = (float4*)out;
        float4 v = o4[i];
        v.x *= inv; v.y *= inv; v.z *= inv; v.w *= inv;
        o4[i] = v;
    }
}

// ---------------------------------------------------------------------------
extern "C" void kernel_launch(void* const* d_in, const int* in_sizes, int n_in,
                              void* d_out, int out_size, void* d_ws, size_t ws_size,
                              hipStream_t stream) {
    (void)in_sizes; (void)n_in; (void)out_size; (void)ws_size;
    const float* A   = (const float*)d_in[0];
    const float* X   = (const float*)d_in[1];
    const float* Ww  = (const float*)d_in[2];
    const float* bw  = (const float*)d_in[3];
    const float* Wfc = (const float*)d_in[4];
    const float* bfc = (const float*)d_in[5];
    float* out = (float*)d_out;

    char* ws = (char*)d_ws;
    unsigned*       hP   = (unsigned*)ws;                                    // 4 MB
    unsigned short* catB = (unsigned short*)(ws + (size_t)NN * DOUT * 2);    // 8 MB
    float* sumsq = (float*)(ws + (size_t)NN * DOUT * 2 + (size_t)NN * DCAT * 2);

    k_h    <<<1024, 256, 0, stream>>>(X, Ww, bw, hP);
    k_castX<<<8192, 256, 0, stream>>>(X, catB);
    k_pool <<<  64, 256, 0, stream>>>(A, hP, catB);
    k_init <<<   1,   1, 0, stream>>>(sumsq);
    k_out  <<<1024, 256, 0, stream>>>(catB, Wfc, bfc, out, sumsq);
    k_scale<<<2048, 256, 0, stream>>>(out, sumsq);
}
